// AlexNetQuant_58935541236428
// MI455X (gfx1250) — 8-bit matrix operand order repaired and checked by reading, compiled, not run on hardware
//
#include <hip/hip_runtime.h>
#include <cstdint>
#include <cstddef>


typedef int v8i __attribute__((ext_vector_type(8)));

__global__ void k_binarize(const float* __restrict__ x, int8_t* __restrict__ y,
                           size_t n) {
  size_t i = (size_t)blockIdx.x * blockDim.x + threadIdx.x;
  if (i >= n) return;
  y[i] = (x[i] >= 0.0f) ? (int8_t)1 : (int8_t)-1;
}

__global__ void k_binw(const float* __restrict__ w, int8_t* __restrict__ A,
                       int Mreal, int K, int Kpad, size_t total) {
  size_t i = (size_t)blockIdx.x * blockDim.x + threadIdx.x;
  if (i >= total) return;
  int m = (int)(i / (size_t)Kpad);
  int k = (int)(i - (size_t)m * Kpad);
  int8_t v = 0;
  if (m < Mreal && k < K) v = (w[(size_t)m * K + k] >= 0.0f) ? 1 : -1;
  A[i] = v;
}

__global__ void k_im2col(const int8_t* __restrict__ in, int8_t* __restrict__ col,
                         int C, int H, int W, int KH, int KW, int stride,
                         int pad, int OH, int OW, int K, int Kpad,
                         size_t total) {
  size_t i = (size_t)blockIdx.x * blockDim.x + threadIdx.x;
  if (i >= total) return;
  int n = (int)(i / (size_t)Kpad);
  int k = (int)(i - (size_t)n * Kpad);
  int8_t v = 0;
  if (k < K) {
    int khkw = KH * KW;
    int c = k / khkw;
    int rem = k - c * khkw;
    int ky = rem / KW;
    int kx = rem - ky * KW;
    int sp = n % (OH * OW);
    int img = n / (OH * OW);
    int oh = sp / OW;
    int ow = sp - oh * OW;
    int ih = oh * stride - pad + ky;
    int iw = ow * stride - pad + kx;
    if (ih >= 0 && ih < H && iw >= 0 && iw < W)
      v = in[(((size_t)img * C + c) * H + ih) * W + iw];
  }
  col[i] = v;
}

__global__ void k_maxpool3s2(const int8_t* __restrict__ in,
                             int8_t* __restrict__ out, int C, int H, int W,
                             int OH, int OW, size_t total) {
  size_t i = (size_t)blockIdx.x * blockDim.x + threadIdx.x;
  if (i >= total) return;
  size_t t = i;
  int ow = (int)(t % OW); t /= OW;
  int oh = (int)(t % OH); t /= OH;
  int c  = (int)(t % C);  t /= C;
  int img = (int)t;
  const int8_t* base = in + (((size_t)img * C + c) * H + oh * 2) * W + ow * 2;
  int m = -128;
#pragma unroll
  for (int dy = 0; dy < 3; ++dy)
#pragma unroll
    for (int dx = 0; dx < 3; ++dx) {
      int v = base[dy * W + dx];
      m = v > m ? v : m;
    }
  out[i] = (int8_t)m;
}

template <int MODE>
__global__ __launch_bounds__(128)
void k_gemm_wmma(const int8_t* __restrict__ A, const int8_t* __restrict__ B,
                 void* __restrict__ out, int M, int N, int Kpad, int spatial,
                 int Mout) {
  const int lane = threadIdx.x & 31;
  const int wave = threadIdx.x >> 5;
  const int wm = wave & 1;
  const int wn = wave >> 1;
  const int m0 = blockIdx.x * 64 + wm * 32;
  const int n0 = blockIdx.y * 128 + wn * 64;
  const int half = lane >> 4;
  const int l16 = lane & 15;

  const int8_t* arow[2];
  const int8_t* brow[4];
#pragma unroll
  for (int ti = 0; ti < 2; ++ti)
    arow[ti] = A + (size_t)(m0 + ti * 16 + l16) * Kpad;
#pragma unroll
  for (int tj = 0; tj < 4; ++tj)
    brow[tj] = B + (size_t)(n0 + tj * 16 + l16) * Kpad;

  v8i acc[2][4];
#pragma unroll
  for (int i = 0; i < 2; ++i)
#pragma unroll
    for (int j = 0; j < 4; ++j)
      acc[i][j] = (v8i){0, 0, 0, 0, 0, 0, 0, 0};

  for (int k0 = 0; k0 < Kpad; k0 += 64) {
    if (k0 + 64 < Kpad) {
#pragma unroll
      for (int ti = 0; ti < 2; ++ti) __builtin_prefetch(arow[ti] + k0 + 64, 0, 1);
#pragma unroll
      for (int tj = 0; tj < 4; ++tj) __builtin_prefetch(brow[tj] + k0 + 64, 0, 1);
    }
    v8i a[2], b[4];
#pragma unroll
    for (int v = 0; v < 8; ++v) {
      int ka = (v >> 2) * 32 + half * 16 + (v & 3) * 4;
#pragma unroll
      for (int ti = 0; ti < 2; ++ti)
        a[ti][v] = *(const int*)(arow[ti] + k0 + ka);
      int kb = (v >> 2) * 32 + half * 16 + (v & 3) * 4;
#pragma unroll
      for (int tj = 0; tj < 4; ++tj)
        b[tj][v] = *(const int*)(brow[tj] + k0 + kb);
    }
#pragma unroll
    for (int ti = 0; ti < 2; ++ti)
#pragma unroll
      for (int tj = 0; tj < 4; ++tj)
        acc[ti][tj] = __builtin_amdgcn_wmma_i32_16x16x64_iu8(
            true, a[ti], true, b[tj], acc[ti][tj], false, false);
  }

#pragma unroll
  for (int ti = 0; ti < 2; ++ti) {
#pragma unroll
    for (int tj = 0; tj < 4; ++tj) {
#pragma unroll
      for (int r = 0; r < 8; ++r) {
        int m = m0 + ti * 16 + half * 8 + r;
        int n = n0 + tj * 16 + l16;
        int v = acc[ti][tj][r];
        if (MODE == 0) {
          int img = n / spatial;
          int sp = n - img * spatial;
          ((int8_t*)out)[((size_t)img * M + m) * spatial + sp] =
              (v >= 1) ? (int8_t)1 : (int8_t)0;
        } else if (MODE == 1) {
          ((int8_t*)out)[(size_t)n * M + m] = (v >= 1) ? (int8_t)1 : (int8_t)0;
        } else {
          if (m < Mout) ((float*)out)[(size_t)n * Mout + m] = (float)v;
        }
      }
    }
  }
}

static inline size_t cdiv_(size_t a, size_t b) { return (a + b - 1) / b; }

extern "C" void kernel_launch(void* const* d_in, const int* in_sizes, int n_in,
                              void* d_out, int out_size, void* d_ws,
                              size_t ws_size, hipStream_t stream) {
  (void)in_sizes; (void)n_in; (void)out_size; (void)ws_size;
  const float* x   = (const float*)d_in[0];
  const float* w1  = (const float*)d_in[1];
  const float* w2  = (const float*)d_in[2];
  const float* w3  = (const float*)d_in[3];
  const float* w4  = (const float*)d_in[4];
  const float* w5  = (const float*)d_in[5];
  const float* fw1 = (const float*)d_in[6];
  const float* fw2 = (const float*)d_in[7];
  const float* fw3 = (const float*)d_in[8];
  float* out = (float*)d_out;

  int8_t* ws = (int8_t*)d_ws;
  size_t off = 0;
  auto carve = [&](size_t bytes) {
    size_t r = off;
    off += (bytes + 255) & ~(size_t)255;
    return r;
  };
  int8_t* xbin = ws + carve(128ull * 3 * 224 * 224);
  int8_t* f0   = ws + carve(24780800ull);
  int8_t* f1   = ws + carve(24780800ull);
  int8_t* col  = ws + carve(149299200ull);
  int8_t* wbuf = ws + carve(37748736ull);

  const int B = 128;
  const int T = 256;
  const int GT = 128;

  {
    size_t n = (size_t)B * 3 * 224 * 224;
    k_binarize<<<(unsigned)cdiv_(n, T), T, 0, stream>>>(x, xbin, n);
  }

  {
    size_t N = (size_t)B * 55 * 55;
    size_t tot = N * 384;
    k_im2col<<<(unsigned)cdiv_(tot, T), T, 0, stream>>>(
        xbin, col, 3, 224, 224, 11, 11, 4, 2, 55, 55, 363, 384, tot);
    size_t wt = 64ull * 384;
    k_binw<<<(unsigned)cdiv_(wt, T), T, 0, stream>>>(w1, wbuf, 64, 363, 384, wt);
    dim3 g(64 / 64, (unsigned)(N / 128));
    k_gemm_wmma<0><<<g, GT, 0, stream>>>(wbuf, col, f0, 64, (int)N, 384,
                                         55 * 55, 0);
  }
  {
    size_t tot = (size_t)B * 64 * 27 * 27;
    k_maxpool3s2<<<(unsigned)cdiv_(tot, T), T, 0, stream>>>(f0, f1, 64, 55, 55,
                                                            27, 27, tot);
  }

  {
    size_t N = (size_t)B * 27 * 27;
    size_t tot = N * 1600;
    k_im2col<<<(unsigned)cdiv_(tot, T), T, 0, stream>>>(
        f1, col, 64, 27, 27, 5, 5, 1, 2, 27, 27, 1600, 1600, tot);
    size_t wt = 192ull * 1600;
    k_binw<<<(unsigned)cdiv_(wt, T), T, 0, stream>>>(w2, wbuf, 192, 1600, 1600, wt);
    dim3 g(192 / 64, (unsigned)(N / 128));
    k_gemm_wmma<0><<<g, GT, 0, stream>>>(wbuf, col, f0, 192, (int)N, 1600,
                                         27 * 27, 0);
  }
  {
    size_t tot = (size_t)B * 192 * 13 * 13;
    k_maxpool3s2<<<(unsigned)cdiv_(tot, T), T, 0, stream>>>(f0, f1, 192, 27, 27,
                                                            13, 13, tot);
  }

  {
    size_t N = (size_t)B * 13 * 13;
    size_t tot = N * 1728;
    k_im2col<<<(unsigned)cdiv_(tot, T), T, 0, stream>>>(
        f1, col, 192, 13, 13, 3, 3, 1, 1, 13, 13, 1728, 1728, tot);
    size_t wt = 384ull * 1728;
    k_binw<<<(unsigned)cdiv_(wt, T), T, 0, stream>>>(w3, wbuf, 384, 1728, 1728, wt);
    dim3 g(384 / 64, (unsigned)(N / 128));
    k_gemm_wmma<0><<<g, GT, 0, stream>>>(wbuf, col, f0, 384, (int)N, 1728,
                                         13 * 13, 0);
  }

  {
    size_t N = (size_t)B * 13 * 13;
    size_t tot = N * 3456;
    k_im2col<<<(unsigned)cdiv_(tot, T), T, 0, stream>>>(
        f0, col, 384, 13, 13, 3, 3, 1, 1, 13, 13, 3456, 3456, tot);
    size_t wt = 256ull * 3456;
    k_binw<<<(unsigned)cdiv_(wt, T), T, 0, stream>>>(w4, wbuf, 256, 3456, 3456, wt);
    dim3 g(256 / 64, (unsigned)(N / 128));
    k_gemm_wmma<0><<<g, GT, 0, stream>>>(wbuf, col, f1, 256, (int)N, 3456,
                                         13 * 13, 0);
  }

  {
    size_t N = (size_t)B * 13 * 13;
    size_t tot = N * 2304;
    k_im2col<<<(unsigned)cdiv_(tot, T), T, 0, stream>>>(
        f1, col, 256, 13, 13, 3, 3, 1, 1, 13, 13, 2304, 2304, tot);
    size_t wt = 256ull * 2304;
    k_binw<<<(unsigned)cdiv_(wt, T), T, 0, stream>>>(w5, wbuf, 256, 2304, 2304, wt);
    dim3 g(256 / 64, (unsigned)(N / 128));
    k_gemm_wmma<0><<<g, GT, 0, stream>>>(wbuf, col, f0, 256, (int)N, 2304,
                                         13 * 13, 0);
  }
  {
    size_t tot = (size_t)B * 256 * 6 * 6;
    k_maxpool3s2<<<(unsigned)cdiv_(tot, T), T, 0, stream>>>(f0, f1, 256, 13, 13,
                                                            6, 6, tot);
  }

  {
    size_t wt = 4096ull * 9216;
    k_binw<<<(unsigned)cdiv_(wt, T), T, 0, stream>>>(fw1, wbuf, 4096, 9216, 9216, wt);
    dim3 g(4096 / 64, 128 / 128);
    k_gemm_wmma<1><<<g, GT, 0, stream>>>(wbuf, f1, f0, 4096, 128, 9216, 0, 0);
  }
  {
    size_t wt = 4096ull * 4096;
    k_binw<<<(unsigned)cdiv_(wt, T), T, 0, stream>>>(fw2, wbuf, 4096, 4096, 4096, wt);
    dim3 g(4096 / 64, 128 / 128);
    k_gemm_wmma<1><<<g, GT, 0, stream>>>(wbuf, f0, f1, 4096, 128, 4096, 0, 0);
  }
  {
    size_t wt = 1024ull * 4096;
    k_binw<<<(unsigned)cdiv_(wt, T), T, 0, stream>>>(fw3, wbuf, 1000, 4096, 4096, wt);
    dim3 g(1024 / 64, 128 / 128);
    k_gemm_wmma<2><<<g, GT, 0, stream>>>(wbuf, f1, out, 1024, 128, 4096, 0, 1000);
  }
}
